// QSARMPNN_11235634447069
// MI455X (gfx1250) — matrix operand order repaired and checked by reading, not compiled, not run on hardware
//
#include <hip/hip_runtime.h>

typedef __attribute__((ext_vector_type(16))) __bf16 v16bf;
typedef __attribute__((ext_vector_type(8)))  float  v8f;

#define NN 25000
#define EE 50000
#define GG 1000
#define NP 25
#define DD 64
#define KTOT 4160          // 4096 (t x h outer) + 64 (eb2 bias rows, t==1)
#define NIT 130            // KTOT / 32
#define FRAG_ELEMS (NIT*4*32*16)   // 266240 bf16
#define FRAG_PAD   2048            // one extra it-slice, loaded by the pipeline but unused

// ---- fragment K mappings (ISA 7.12.2, 16-bit operands, wave32) ----
// A-matrix 16x32: lanes 0-15 hold K 0..15; lanes 16-31 hold K 16..31 (contiguous, matches B)
__device__ __forceinline__ int kmapA(int v, int lane) {
    return v + (lane & 16);
}
// B-matrix 32x16: lanes 0-15 hold K 0..15; lanes 16-31 hold K 16..31 (contiguous)
__device__ __forceinline__ int kmapB(int v, int lane) {
    return v + (lane & 16);
}

__device__ __forceinline__ float sigf(float x) { return 1.0f / (1.0f + __expf(-x)); }

// ---------------- node projection: h = relu(n_feats @ proj_W + b) ----------------
__global__ __launch_bounds__(256) void proj_kernel(
    const float* __restrict__ nf, const float* __restrict__ W,
    const float* __restrict__ b, float* __restrict__ h)
{
    int idx = blockIdx.x * 256 + threadIdx.x;
    if (idx >= NN * DD) return;
    int n = idx >> 6, d = idx & 63;
    const float* x = nf + (size_t)n * 74;
    float s = b[d];
#pragma unroll
    for (int k = 0; k < 74; ++k) s += x[k] * W[k * DD + d];
    h[idx] = fmaxf(s, 0.0f);
}

// ---------------- edge hidden: t = relu(e_feats @ eW1 + eb1) ----------------
__global__ __launch_bounds__(256) void ehid_kernel(
    const float* __restrict__ ef, const float* __restrict__ W,
    const float* __restrict__ b, float* __restrict__ t)
{
    int idx = blockIdx.x * 256 + threadIdx.x;
    if (idx >= EE * DD) return;
    int e = idx >> 6, d = idx & 63;
    const float* x = ef + (size_t)e * 12;
    float s = b[d];
#pragma unroll
    for (int k = 0; k < 12; ++k) s += x[k] * W[k * DD + d];
    t[idx] = fmaxf(s, 0.0f);
}

// -------- pack W' = [eW2 viewed as (4096,64); eb2 as (64,64)] into bf16 B-fragments --------
__global__ __launch_bounds__(256) void wfrag_kernel(
    const float* __restrict__ eW2, const float* __restrict__ eb2, __bf16* __restrict__ Wf)
{
    int idx = blockIdx.x * 256 + threadIdx.x;
    if (idx >= FRAG_ELEMS) return;
    int v    = idx & 15;
    int lane = (idx >> 4) & 31;
    int nt   = (idx >> 9) & 3;
    int it   = idx >> 11;
    int kg = it * 32 + kmapB(v, lane);
    int n  = nt * 16 + (lane & 15);
    float val = (kg < 4096) ? eW2[(size_t)kg * 64 + n]
                            : eb2[(size_t)(kg - 4096) * 64 + n];
    Wf[idx] = (__bf16)val;
}

// -------- WMMA message pass: msg = X @ W', X_e[r] = t_e[r>>6]*h_src[r&63]; atomic segment-sum --------
__global__ __launch_bounds__(256) void msg_kernel(
    const float* __restrict__ h, const float* __restrict__ t,
    const __bf16* __restrict__ Wf, const int* __restrict__ src,
    const int* __restrict__ dst, float* __restrict__ mbuf)
{
    __shared__ float t_lds[8 * 16 * 65];   // 16 edges x 64, stride 65 (bank-spread), per wave
    int lane = threadIdx.x & 31;
    int wid  = threadIdx.x >> 5;
    int eb0  = blockIdx.x * 128 + wid * 16;
    float* tl = &t_lds[wid * 16 * 65];

    // stage this wave's 16 edges of t into LDS (zero for tail edges)
    for (int i = lane; i < 16 * 64; i += 32) {
        int m = i >> 6, c = i & 63;
        int e = eb0 + m;
        tl[m * 65 + c] = (e < EE) ? t[(size_t)e * 64 + c] : 0.0f;
    }
    __syncthreads();

    // preload the 32 h[src] values this lane's A-fragment slots ever need
    int mrow = lane & 15;
    int em = eb0 + mrow;
    int ec = em < EE ? em : (EE - 1);
    const float* hp = h + (size_t)src[ec] * 64;
    float hlo[16], hhi[16];
#pragma unroll
    for (int v = 0; v < 16; ++v) {
        int c = kmapA(v, lane);
        hlo[v] = hp[c];
        hhi[v] = hp[c + 32];
    }

    v8f acc[4];
#pragma unroll
    for (int nt = 0; nt < 4; ++nt)
#pragma unroll
        for (int g = 0; g < 8; ++g) acc[nt][g] = 0.0f;

    // ---- software-pipelined K loop: double-buffered B fragments ----
    const __bf16* wb = Wf + (size_t)lane * 16;
    v16bf bcur[4], bnext[4];
#pragma unroll
    for (int nt = 0; nt < 4; ++nt)
        bcur[nt] = *reinterpret_cast<const v16bf*>(wb + nt * 512);

    auto step = [&](int it, float tv, const float* hsel) {
        // issue next iteration's B loads (pad slice makes it+1 always valid)
        const __bf16* wn = wb + (size_t)(it + 1) * 2048;
        __builtin_prefetch(wn + 4096, 0, 1);            // stream 2 slices ahead
#pragma unroll
        for (int nt = 0; nt < 4; ++nt)
            bnext[nt] = *reinterpret_cast<const v16bf*>(wn + nt * 512);
        v16bf a;
#pragma unroll
        for (int v = 0; v < 16; ++v) a[v] = (__bf16)(tv * hsel[v]);
#pragma unroll
        for (int nt = 0; nt < 4; ++nt)
            acc[nt] = __builtin_amdgcn_wmma_f32_16x16x32_bf16(
                false, a, false, bcur[nt], (short)0, acc[nt], false, false);
#pragma unroll
        for (int nt = 0; nt < 4; ++nt) bcur[nt] = bnext[nt];
    };

    const float* trow = tl + mrow * 65;
    for (int kc = 0; kc < 64; ++kc) {      // branch-free main loop: one t value per 2 K-steps
        float tv = trow[kc];
        step(2 * kc,     tv, hlo);
        step(2 * kc + 1, tv, hhi);
    }
    step(128, 1.0f, hlo);                  // peeled eb2 bias rows (t == 1)
    step(129, 1.0f, hhi);

    // scatter-accumulate into per-destination message buffer (segment_sum over dst)
    int coll  = lane & 15;
    int rbase = (lane < 16) ? 0 : 8;
#pragma unroll
    for (int g = 0; g < 8; ++g) {
        int e = eb0 + rbase + g;
        if (e < EE) {
            int dn = dst[e];
#pragma unroll
            for (int nt = 0; nt < 4; ++nt)
                atomicAdd(&mbuf[(size_t)dn * 64 + nt * 16 + coll], acc[nt][g]);
        }
    }
}

// ---------------- fused relu(m + conv_b) + GRU cell, hidden updated in place ----------------
__global__ __launch_bounds__(256) void gru_kernel(
    const float* __restrict__ mbuf, const float* __restrict__ conv_b,
    float* __restrict__ g, const float* __restrict__ Wi, const float* __restrict__ Wh,
    const float* __restrict__ bi, const float* __restrict__ bh)
{
    __shared__ float xm[4][64], xg[4][64];
    int j  = threadIdx.x & 63;
    int sl = threadIdx.x >> 6;
    int n  = blockIdx.x * 4 + sl;
    int nc = n < NN ? n : NN - 1;
    xm[sl][j] = fmaxf(mbuf[(size_t)nc * 64 + j] + conv_b[j], 0.0f);
    xg[sl][j] = g[(size_t)nc * 64 + j];
    __syncthreads();
    float ir = bi[j],       hr = bh[j];
    float iz = bi[64 + j],  hz = bh[64 + j];
    float in_ = bi[128 + j], hn = bh[128 + j];
    for (int i = 0; i < 64; ++i) {
        float mi = xm[sl][i], gi = xg[sl][i];
        ir  += mi * Wi[i * 192 + j];        hr += gi * Wh[i * 192 + j];
        iz  += mi * Wi[i * 192 + 64 + j];   hz += gi * Wh[i * 192 + 64 + j];
        in_ += mi * Wi[i * 192 + 128 + j];  hn += gi * Wh[i * 192 + 128 + j];
    }
    float r  = sigf(ir + hr);
    float z  = sigf(iz + hz);
    float nn = tanhf(in_ + r * hn);
    if (n < NN) g[(size_t)n * 64 + j] = (1.0f - z) * nn + z * xg[sl][j];
}

// ---------------- Set2Set 3-layer LSTM step: one block (64 threads) per graph ----------------
__global__ __launch_bounds__(64) void lstm_kernel(
    const float* __restrict__ q_star,
    const float* __restrict__ lWi0, const float* __restrict__ lWh0, const float* __restrict__ lb0,
    const float* __restrict__ lWi1, const float* __restrict__ lWh1, const float* __restrict__ lb1,
    const float* __restrict__ lWi2, const float* __restrict__ lWh2, const float* __restrict__ lb2,
    float* __restrict__ hsb, float* __restrict__ csb, float* __restrict__ q)
{
    __shared__ float x[128], hb[64];
    int gph = blockIdx.x;
    int j = threadIdx.x;
    x[j]      = q_star[(size_t)gph * 128 + j];
    x[64 + j] = q_star[(size_t)gph * 128 + 64 + j];
    for (int l = 0; l < 3; ++l) {
        const float* Wi = (l == 0) ? lWi0 : (l == 1) ? lWi1 : lWi2;
        const float* Wh = (l == 0) ? lWh0 : (l == 1) ? lWh1 : lWh2;
        const float* b  = (l == 0) ? lb0  : (l == 1) ? lb1  : lb2;
        int xdim = (l == 0) ? 128 : 64;
        float* hsl = hsb + ((size_t)l * GG + gph) * 64;
        float* csl = csb + ((size_t)l * GG + gph) * 64;
        hb[j] = hsl[j];
        __syncthreads();
        float gi = b[j], gf = b[64 + j], ggt = b[128 + j], go = b[192 + j];
        for (int i = 0; i < xdim; ++i) {
            float xi = x[i];
            gi  += xi * Wi[i * 256 + j];
            gf  += xi * Wi[i * 256 + 64 + j];
            ggt += xi * Wi[i * 256 + 128 + j];
            go  += xi * Wi[i * 256 + 192 + j];
        }
        for (int i = 0; i < 64; ++i) {
            float hi = hb[i];
            gi  += hi * Wh[i * 256 + j];
            gf  += hi * Wh[i * 256 + 64 + j];
            ggt += hi * Wh[i * 256 + 128 + j];
            go  += hi * Wh[i * 256 + 192 + j];
        }
        float c    = sigf(gf) * csl[j] + sigf(gi) * tanhf(ggt);
        float hnew = sigf(go) * tanhf(c);
        csl[j] = c;
        hsl[j] = hnew;
        __syncthreads();
        x[j] = hnew;
        __syncthreads();
    }
    q[(size_t)gph * 64 + j] = x[j];
}

// ---------------- attention/readout: one wave per graph (25 nodes, D=64) ----------------
__global__ __launch_bounds__(32) void attn_kernel(
    const float* __restrict__ h, const float* __restrict__ q, float* __restrict__ q_star)
{
    int gph  = blockIdx.x;
    int lane = threadIdx.x;
    float q0 = q[(size_t)gph * 64 + lane];
    float q1 = q[(size_t)gph * 64 + lane + 32];
    float ev[NP];
#pragma unroll
    for (int nl = 0; nl < NP; ++nl) {
        const float* hp = h + ((size_t)gph * NP + nl) * 64;
        float d = hp[lane] * q0 + hp[lane + 32] * q1;
#pragma unroll
        for (int s = 16; s >= 1; s >>= 1) d += __shfl_xor(d, s, 32);
        ev[nl] = d;
    }
    float mx = ev[0];
#pragma unroll
    for (int nl = 1; nl < NP; ++nl) mx = fmaxf(mx, ev[nl]);
    float ssum = 0.0f;
#pragma unroll
    for (int nl = 0; nl < NP; ++nl) { ev[nl] = __expf(ev[nl] - mx); ssum += ev[nl]; }
    float inv = 1.0f / ssum;
    float r0 = 0.0f, r1 = 0.0f;
#pragma unroll
    for (int nl = 0; nl < NP; ++nl) {
        const float* hp = h + ((size_t)gph * NP + nl) * 64;
        float a = ev[nl] * inv;
        r0 += a * hp[lane];
        r1 += a * hp[lane + 32];
    }
    q_star[(size_t)gph * 128 + lane]      = q0;
    q_star[(size_t)gph * 128 + 32 + lane] = q1;
    q_star[(size_t)gph * 128 + 64 + lane] = r0;
    q_star[(size_t)gph * 128 + 96 + lane] = r1;
}

// ---------------- head: sigmoid(relu(q_star @ pW + pb) @ fW + fb) ----------------
__global__ __launch_bounds__(64) void predict_kernel(
    const float* __restrict__ q_star, const float* __restrict__ pW, const float* __restrict__ pb,
    const float* __restrict__ fW, const float* __restrict__ fb, float* __restrict__ out)
{
    __shared__ float red[64];
    int gph = blockIdx.x, j = threadIdx.x;
    const float* x = q_star + (size_t)gph * 128;
    float s = pb[j];
    for (int i = 0; i < 128; ++i) s += x[i] * pW[i * 64 + j];
    red[j] = fmaxf(s, 0.0f) * fW[j];
    __syncthreads();
    if (j == 0) {
        float tt = fb[0];
        for (int i = 0; i < 64; ++i) tt += red[i];
        out[gph] = sigf(tt);
    }
}

extern "C" void kernel_launch(void* const* d_in, const int* in_sizes, int n_in,
                              void* d_out, int out_size, void* d_ws, size_t ws_size,
                              hipStream_t stream)
{
    const float* n_feats = (const float*)d_in[0];
    const float* e_feats = (const float*)d_in[1];
    const int*   src     = (const int*)d_in[2];
    const int*   dst     = (const int*)d_in[3];
    const float* proj_W  = (const float*)d_in[5];
    const float* proj_b  = (const float*)d_in[6];
    const float* eW1     = (const float*)d_in[7];
    const float* eb1     = (const float*)d_in[8];
    const float* eW2     = (const float*)d_in[9];
    const float* eb2     = (const float*)d_in[10];
    const float* conv_b  = (const float*)d_in[11];
    const float* gru_Wi  = (const float*)d_in[12];
    const float* gru_Wh  = (const float*)d_in[13];
    const float* gru_bi  = (const float*)d_in[14];
    const float* gru_bh  = (const float*)d_in[15];
    const float* lWi0 = (const float*)d_in[16];
    const float* lWh0 = (const float*)d_in[17];
    const float* lb0  = (const float*)d_in[18];
    const float* lWi1 = (const float*)d_in[19];
    const float* lWh1 = (const float*)d_in[20];
    const float* lb1  = (const float*)d_in[21];
    const float* lWi2 = (const float*)d_in[22];
    const float* lWh2 = (const float*)d_in[23];
    const float* lb2  = (const float*)d_in[24];
    const float* pW = (const float*)d_in[25];
    const float* pb = (const float*)d_in[26];
    const float* fW = (const float*)d_in[27];
    const float* fb = (const float*)d_in[28];

    // workspace carve (all chunk sizes multiples of 256 B)
    char* ws = (char*)d_ws;
    float*  h      = (float*)ws;                                   // NN*64
    float*  t      = h + (size_t)NN * 64;                          // EE*64
    float*  mb     = t + (size_t)EE * 64;                          // NN*64
    __bf16* Wf     = (__bf16*)(mb + (size_t)NN * 64);              // FRAG_ELEMS (+pad) bf16
    float*  q_star = (float*)((char*)Wf +
                     (((size_t)(FRAG_ELEMS + FRAG_PAD) * 2 + 255) & ~(size_t)255));
    float*  q      = q_star + (size_t)GG * 128;
    float*  hsb    = q + (size_t)GG * 64;                          // 3*GG*64
    float*  csb    = hsb + (size_t)3 * GG * 64;                    // 3*GG*64

    proj_kernel<<<(NN * DD + 255) / 256, 256, 0, stream>>>(n_feats, proj_W, proj_b, h);
    ehid_kernel<<<(EE * DD + 255) / 256, 256, 0, stream>>>(e_feats, eW1, eb1, t);
    wfrag_kernel<<<(FRAG_ELEMS + 255) / 256, 256, 0, stream>>>(eW2, eb2, Wf);

    hipMemsetAsync(q_star, 0, (size_t)GG * 128 * sizeof(float), stream);
    hipMemsetAsync(hsb,    0, (size_t)3 * GG * 64 * sizeof(float), stream);
    hipMemsetAsync(csb,    0, (size_t)3 * GG * 64 * sizeof(float), stream);

    for (int r = 0; r < 3; ++r) {
        hipMemsetAsync(mb, 0, (size_t)NN * 64 * sizeof(float), stream);
        msg_kernel<<<(EE + 127) / 128, 256, 0, stream>>>(h, t, Wf, src, dst, mb);
        gru_kernel<<<(NN + 3) / 4, 256, 0, stream>>>(mb, conv_b, h, gru_Wi, gru_Wh, gru_bi, gru_bh);
    }

    for (int i = 0; i < 3; ++i) {
        lstm_kernel<<<GG, 64, 0, stream>>>(q_star, lWi0, lWh0, lb0, lWi1, lWh1, lb1,
                                           lWi2, lWh2, lb2, hsb, csb, q);
        attn_kernel<<<GG, 32, 0, stream>>>(h, q, q_star);
    }

    predict_kernel<<<GG, 64, 0, stream>>>(q_star, pW, pb, fW, fb, (float*)d_out);
}
